// MultiHeadAttention_1898375545047
// MI455X (gfx1250) — matrix operand order repaired and checked by reading, not compiled, not run on hardware
//
#include <hip/hip_runtime.h>
#include <hip/hip_bf16.h>

// ---------------------------------------------------------------------------
// MultiHeadAttention for MI455X (gfx1250), wave32 + v_wmma_f32_16x16x32_f16.
// Bandwidth-bound (~270MB HBM @ 23.3TB/s ~ 12us; ~30 GFLOP is negligible on
// the WMMA units).  All GEMMs run through WMMA with f16 operands / f32
// accumulation; intermediates in f16; shared tiles staged in LDS (320KB/WGP);
// fragment loads software-pipelined so WMMAs overlap outstanding loads.
// ---------------------------------------------------------------------------

typedef __attribute__((ext_vector_type(16))) _Float16 v16h;
typedef __attribute__((ext_vector_type(8)))  _Float16 v8h;
typedef __attribute__((ext_vector_type(8)))  float    v8f;

#define N_HEAD 8
#define DK 64
#define BATCH 16
#define SEQ 512
#define DMODEL 512
#define ROWS (BATCH * SEQ)          // 8192
#define KSTEPS (DMODEL / 32)        // 16

// ---- WMMA helpers (layouts per CDNA5 ISA §7.12.2) ----------------

__device__ __forceinline__ v8f wmma32(v16h a, v16h b, v8f c) {
  // 8 args: (neg_a, A, neg_b, B, c_mod, C, reuse_a, reuse_b)
  return __builtin_amdgcn_wmma_f32_16x16x32_f16(false, a, false, b, (short)0, c,
                                                false, false);
}

// A fragment 16x32 f16: lane kh = lane>>4 holds K = kh*8+0..7 (halves 0..7)
// and K = 16+kh*8+0..7 (halves 8..15).  p must point at &A[row][k0 + kh*8].
__device__ __forceinline__ v16h load_a16(const _Float16* __restrict__ p) {
  v8h lo = *(const v8h*)p;
  v8h hi = *(const v8h*)(p + 16);
  v16h a;
#pragma unroll
  for (int i = 0; i < 8; ++i) { a[i] = lo[i]; a[8 + i] = hi[i]; }
  return a;
}

// B fragment 32x16: lane holds column (lane&15), K = kh*8+0..7 (halves 0..7)
// and K = 16+kh*8+0..7 (halves 8..15), same as A.  With B stored transposed
// (Bt[n][k] row-major):  p = &Bt[col][k0 + kh*8]
__device__ __forceinline__ v16h load_b16(const _Float16* __restrict__ p) {
  v8h lo = *(const v8h*)p;
  v8h hi = *(const v8h*)(p + 16);
  v16h b;
#pragma unroll
  for (int i = 0; i < 8; ++i) { b[i] = lo[i]; b[8 + i] = hi[i]; }
  return b;
}

// ---- small prep kernels -----------------------------------------------------

__global__ void __launch_bounds__(256)
cast_f32_to_f16(const float* __restrict__ src, _Float16* __restrict__ dst, int n) {
  int i = blockIdx.x * 256 + threadIdx.x;
  if (i < n) dst[i] = (_Float16)src[i];
}

// LayerNorm over 512 and cast to f16.  One wave per row, 16 elems/lane.
__global__ void __launch_bounds__(256)
ln_cast(const float* __restrict__ x, const float* __restrict__ g,
        const float* __restrict__ bt, _Float16* __restrict__ out) {
  const int lane = threadIdx.x & 31;
  const int row  = blockIdx.x * 8 + (threadIdx.x >> 5);
  const float* xr = x + (size_t)row * DMODEL + lane * 16;
  v8f a = *(const v8f*)xr;
  v8f b = *(const v8f*)(xr + 8);
  float s = 0.f, s2 = 0.f;
#pragma unroll
  for (int i = 0; i < 8; ++i) {
    s  += a[i] + b[i];
    s2 += a[i] * a[i] + b[i] * b[i];
  }
#pragma unroll
  for (int d = 16; d >= 1; d >>= 1) {
    s  += __shfl_xor(s,  d, 32);
    s2 += __shfl_xor(s2, d, 32);
  }
  const float mean = s * (1.f / DMODEL);
  const float var  = s2 * (1.f / DMODEL) - mean * mean;
  const float inv  = rsqrtf(var + 1e-5f);
  const float* gr = g  + lane * 16;
  const float* br = bt + lane * 16;
  v8f g0 = *(const v8f*)gr, g1 = *(const v8f*)(gr + 8);
  v8f b0 = *(const v8f*)br, b1 = *(const v8f*)(br + 8);
  v16h o;
#pragma unroll
  for (int i = 0; i < 8; ++i) {
    o[i]     = (_Float16)((a[i] - mean) * inv * g0[i] + b0[i]);
    o[8 + i] = (_Float16)((b[i] - mean) * inv * g1[i] + b1[i]);
  }
  *(v16h*)(out + (size_t)row * DMODEL + lane * 16) = o;
}

// ---- generic 8192x512x512 WMMA GEMM:  Y = X * Bt^T  -------------------------
// X:  [8192,512] f16 row-major (A matrix).  All 8 waves of a block share the
//     same 16-row A tile -> staged once in LDS (16KB).
// Bt: [512,512]  f16 row-major, Bt[n][k]  (== torch Linear weight W[out,in])
// MODE 0: Yp = f16 [8192,512]
// MODE 1: Yp = f16 [8192,512]  and Yt = f16 [B,H,D,SEQ] transposed (V path)
// MODE 2: Yp = f32 [8192,512]
// Grid: 512 blocks x 256 threads; each wave owns a 16x64 tile.
template <int MODE>
__global__ void __launch_bounds__(256)
gemm_f16(const _Float16* __restrict__ X, const _Float16* __restrict__ Bt,
         void* __restrict__ Yp, _Float16* __restrict__ Yt) {
  __shared__ _Float16 As[16 * DMODEL];          // 16KB
  const int tid  = threadIdx.x;
  const int lane = tid & 31;
  const int kh   = lane >> 4;
  const int l15  = lane & 15;
  const int mbase = blockIdx.x << 4;
  const int nbase = (tid >> 5) << 6;
  // stage shared A tile: 256 threads x 32 halves (64B)
  {
    const int r = tid >> 4, c = (tid & 15) << 5;
    const _Float16* src = X + (size_t)(mbase + r) * DMODEL + c;
    *(v16h*)(As + r * DMODEL + c)      = *(const v16h*)src;
    *(v16h*)(As + r * DMODEL + c + 16) = *(const v16h*)(src + 16);
  }
  __syncthreads();

  const _Float16* ar = As + l15 * DMODEL + kh * 8;
  const _Float16* br = Bt + (size_t)(nbase + l15) * DMODEL + kh * 8;
  v8f c0 = {}, c1 = {}, c2 = {}, c3 = {};
  // software-pipelined K loop: next fragments load while current WMMAs issue
  v16h a  = load_a16(ar);
  v16h b0 = load_b16(br);
  v16h b1 = load_b16(br + 16 * DMODEL);
  v16h b2 = load_b16(br + 32 * DMODEL);
  v16h b3 = load_b16(br + 48 * DMODEL);
#pragma unroll 4
  for (int ks = 1; ks < KSTEPS; ++ks) {
    const int k0 = ks * 32;
    v16h an  = load_a16(ar + k0);
    v16h bn0 = load_b16(br + k0);
    v16h bn1 = load_b16(br + 16 * DMODEL + k0);
    v16h bn2 = load_b16(br + 32 * DMODEL + k0);
    v16h bn3 = load_b16(br + 48 * DMODEL + k0);
    c0 = wmma32(a, b0, c0);
    c1 = wmma32(a, b1, c1);
    c2 = wmma32(a, b2, c2);
    c3 = wmma32(a, b3, c3);
    a = an; b0 = bn0; b1 = bn1; b2 = bn2; b3 = bn3;
  }
  c0 = wmma32(a, b0, c0);
  c1 = wmma32(a, b1, c1);
  c2 = wmma32(a, b2, c2);
  c3 = wmma32(a, b3, c3);

#pragma unroll
  for (int v = 0; v < 8; ++v) {
    const int m = mbase + v + 8 * kh;
    float vals[4] = {c0[v], c1[v], c2[v], c3[v]};
#pragma unroll
    for (int j = 0; j < 4; ++j) {
      const int n = nbase + j * 16 + l15;
      if (MODE == 2) {
        ((float*)Yp)[(size_t)m * DMODEL + n] = vals[j];
      } else {
        ((_Float16*)Yp)[(size_t)m * DMODEL + n] = (_Float16)vals[j];
        if (MODE == 1) {
          const int bb = m >> 9, l = m & 511, hh = n >> 6, d = n & 63;
          Yt[(((size_t)(bb * N_HEAD + hh) * DK + d) << 9) + l] = (_Float16)vals[j];
        }
      }
    }
  }
}

// ---- attention scores: S = Q K^T / 8, diag -> -1e32, into attn[h,b,q,k] ----
// Qh/Kh: f16 [B*SEQ, 512] (head h occupies columns h*64..h*64+63)
// Grid: 4096 blocks (bh * 32 qtiles) x 256 threads; wave -> 16q x 64k tile.
// K-depth is only 64 -> fully unrolled, all 10 fragments loaded before WMMAs.
__global__ void __launch_bounds__(256)
scores_k(const _Float16* __restrict__ Qh, const _Float16* __restrict__ Kh,
         float* __restrict__ attn) {
  const int tid  = threadIdx.x;
  const int lane = tid & 31;
  const int kh   = lane >> 4;
  const int l15  = lane & 15;
  const int bh = blockIdx.x >> 5;       // bh = h*16 + b (torch attn layout)
  const int qt = blockIdx.x & 31;
  const int h = bh >> 4, b = bh & 15;
  const int nbase = (tid >> 5) << 6;    // key-column tile per wave
  const int qbase = qt << 4;
  const _Float16* qp = Qh + (size_t)(b * SEQ + qbase + l15) * DMODEL + h * DK + kh * 8;
  const _Float16* kp = Kh + (size_t)(b * SEQ + nbase + l15) * DMODEL + h * DK + kh * 8;
  v16h a0  = load_a16(qp);
  v16h a1  = load_a16(qp + 32);
  v16h b00 = load_b16(kp);
  v16h b01 = load_b16(kp + 16 * DMODEL);
  v16h b02 = load_b16(kp + 32 * DMODEL);
  v16h b03 = load_b16(kp + 48 * DMODEL);
  v16h b10 = load_b16(kp + 32);
  v16h b11 = load_b16(kp + 16 * DMODEL + 32);
  v16h b12 = load_b16(kp + 32 * DMODEL + 32);
  v16h b13 = load_b16(kp + 48 * DMODEL + 32);
  v8f c0 = {}, c1 = {}, c2 = {}, c3 = {};
  c0 = wmma32(a0, b00, c0);
  c1 = wmma32(a0, b01, c1);
  c2 = wmma32(a0, b02, c2);
  c3 = wmma32(a0, b03, c3);
  c0 = wmma32(a1, b10, c0);
  c1 = wmma32(a1, b11, c1);
  c2 = wmma32(a1, b12, c2);
  c3 = wmma32(a1, b13, c3);
#pragma unroll
  for (int v = 0; v < 8; ++v) {
    const int q = qbase + v + 8 * kh;
    float* dst = attn + ((size_t)bh * SEQ + q) * SEQ;
    float vals[4] = {c0[v], c1[v], c2[v], c3[v]};
#pragma unroll
    for (int j = 0; j < 4; ++j) {
      const int n = nbase + j * 16 + l15;
      float s = vals[j] * 0.125f;            // 1/sqrt(64)
      if (q == n) s = -1e32f;                // diagonal mask
      dst[n] = s;
    }
  }
}

// ---- row softmax in place: 65536 rows of 512, one wave per row --------------
__global__ void __launch_bounds__(256)
softmax_k(float* __restrict__ attn) {
  const int lane = threadIdx.x & 31;
  const size_t row = (size_t)blockIdx.x * 8 + (threadIdx.x >> 5);
  float* p = attn + row * SEQ + lane * 16;
  v8f a = *(v8f*)p;
  v8f b = *(v8f*)(p + 8);
  float m = -3.4e38f;
#pragma unroll
  for (int i = 0; i < 8; ++i) { m = fmaxf(m, a[i]); m = fmaxf(m, b[i]); }
#pragma unroll
  for (int d = 16; d >= 1; d >>= 1) m = fmaxf(m, __shfl_xor(m, d, 32));
  v8f ea, eb;
  float s = 0.f;
#pragma unroll
  for (int i = 0; i < 8; ++i) {
    ea[i] = __expf(a[i] - m);
    eb[i] = __expf(b[i] - m);
    s += ea[i] + eb[i];
  }
#pragma unroll
  for (int d = 16; d >= 1; d >>= 1) s += __shfl_xor(s, d, 32);
  const float inv = 1.f / s;
#pragma unroll
  for (int i = 0; i < 8; ++i) { ea[i] *= inv; eb[i] *= inv; }
  *(v8f*)p       = ea;
  *(v8f*)(p + 8) = eb;
}

// ---- O = P * V: per (b,h) 512x512x64, P f32 -> f16 in regs ------------------
// Vt: f16 [B,H,D,SEQ].  All 8 waves of a block share one (b,h) -> 64x512 V
// tile staged once in LDS (64KB of the WGP's 320KB).
// Oh: f16 [B*SEQ, 512] (head h at columns h*64..) for the FC1 GEMM.
__global__ void __launch_bounds__(256)
pv_k(const float* __restrict__ attn, const _Float16* __restrict__ Vt,
     _Float16* __restrict__ Oh) {
  __shared__ _Float16 Vs[DK * SEQ];             // 64KB
  const int tid  = threadIdx.x;
  const int lane = tid & 31;
  const int kh   = lane >> 4;
  const int l15  = lane & 15;
  const int bh = blockIdx.x >> 2;               // h*16 + b
  const int qt = ((blockIdx.x & 3) << 3) + (tid >> 5);
  const int h = bh >> 4, b = bh & 15;
  const int qbase = qt << 4;
  const _Float16* vb = Vt + (size_t)(b * N_HEAD + h) * DK * SEQ;
  // stage shared V tile: 256 threads x 128 halves (256B)
  {
    const int r = tid >> 2, c = (tid & 3) << 7;
    const _Float16* src = vb + (size_t)r * SEQ + c;
    _Float16* dst = Vs + r * SEQ + c;
#pragma unroll
    for (int j = 0; j < 8; ++j)
      *(v16h*)(dst + j * 16) = *(const v16h*)(src + j * 16);
  }
  __syncthreads();

  const float* pr = attn + ((size_t)bh * SEQ + qbase + l15) * SEQ + kh * 8;
  const _Float16* vr = Vs + (size_t)l15 * SEQ + kh * 8;
  v8f c0 = {}, c1 = {}, c2 = {}, c3 = {};
  // software-pipelined: probability fragments stream from global (L2),
  // V fragments from LDS.
  v8f lo = *(const v8f*)pr;
  v8f hi = *(const v8f*)(pr + 16);
  v16h b0 = load_b16(vr);
  v16h b1 = load_b16(vr + 16 * SEQ);
  v16h b2 = load_b16(vr + 32 * SEQ);
  v16h b3 = load_b16(vr + 48 * SEQ);
#pragma unroll 4
  for (int ks = 1; ks < KSTEPS; ++ks) {
    const int k0 = ks * 32;
    v8f lon = *(const v8f*)(pr + k0);
    v8f hin = *(const v8f*)(pr + k0 + 16);
    v16h bn0 = load_b16(vr + k0);
    v16h bn1 = load_b16(vr + 16 * SEQ + k0);
    v16h bn2 = load_b16(vr + 32 * SEQ + k0);
    v16h bn3 = load_b16(vr + 48 * SEQ + k0);
    v16h a;
#pragma unroll
    for (int i = 0; i < 8; ++i) { a[i] = (_Float16)lo[i]; a[8 + i] = (_Float16)hi[i]; }
    c0 = wmma32(a, b0, c0);
    c1 = wmma32(a, b1, c1);
    c2 = wmma32(a, b2, c2);
    c3 = wmma32(a, b3, c3);
    lo = lon; hi = hin; b0 = bn0; b1 = bn1; b2 = bn2; b3 = bn3;
  }
  {
    v16h a;
#pragma unroll
    for (int i = 0; i < 8; ++i) { a[i] = (_Float16)lo[i]; a[8 + i] = (_Float16)hi[i]; }
    c0 = wmma32(a, b0, c0);
    c1 = wmma32(a, b1, c1);
    c2 = wmma32(a, b2, c2);
    c3 = wmma32(a, b3, c3);
  }
#pragma unroll
  for (int v = 0; v < 8; ++v) {
    const int m = qbase + v + 8 * kh;
    _Float16* dst = Oh + (size_t)(b * SEQ + m) * DMODEL + h * DK;
    float vals[4] = {c0[v], c1[v], c2[v], c3[v]};
#pragma unroll
    for (int j = 0; j < 4; ++j) dst[j * 16 + l15] = (_Float16)vals[j];
  }
}

// ---------------------------------------------------------------------------

extern "C" void kernel_launch(void* const* d_in, const int* in_sizes, int n_in,
                              void* d_out, int out_size, void* d_ws, size_t ws_size,
                              hipStream_t stream) {
  const float* q    = (const float*)d_in[0];
  const float* k    = (const float*)d_in[1];
  const float* v    = (const float*)d_in[2];
  const float* Wq   = (const float*)d_in[3];
  const float* Wk   = (const float*)d_in[4];
  const float* Wv   = (const float*)d_in[5];
  const float* fc1  = (const float*)d_in[6];
  const float* fc2  = (const float*)d_in[7];
  const float* ln1g = (const float*)d_in[8];
  const float* ln1b = (const float*)d_in[9];
  const float* ln2g = (const float*)d_in[10];
  const float* ln2b = (const float*)d_in[11];
  const float* ln3g = (const float*)d_in[12];
  const float* ln3b = (const float*)d_in[13];

  float* out_dyn  = (float*)d_out;                       // [8192,512]
  float* out_stat = out_dyn + (size_t)ROWS * DMODEL;     // [8192,512]
  float* attn     = out_dyn + 2 * (size_t)ROWS * DMODEL; // [128,512,512]

  // workspace layout (bytes); total ~53 MB
  char* ws = (char*)d_ws;
  const size_t WSZ  = (size_t)DMODEL * DMODEL * sizeof(_Float16); // 512KB
  const size_t XSZ  = (size_t)ROWS * DMODEL * sizeof(_Float16);   // 8MB
  _Float16* WqH = (_Float16*)(ws);
  _Float16* WkH = (_Float16*)(ws + 1 * WSZ);
  _Float16* WvH = (_Float16*)(ws + 2 * WSZ);
  _Float16* W1H = (_Float16*)(ws + 3 * WSZ);
  _Float16* W2H = (_Float16*)(ws + 4 * WSZ);
  _Float16* XH  = (_Float16*)(ws + 5 * WSZ);
  _Float16* QH  = (_Float16*)(ws + 5 * WSZ + 1 * XSZ);
  _Float16* KH  = (_Float16*)(ws + 5 * WSZ + 2 * XSZ);
  _Float16* VH  = (_Float16*)(ws + 5 * WSZ + 3 * XSZ);
  _Float16* VT  = (_Float16*)(ws + 5 * WSZ + 4 * XSZ);
  _Float16* OH  = (_Float16*)(ws + 5 * WSZ + 5 * XSZ);

  const int wn = DMODEL * DMODEL;                // 262144
  const dim3 cblk(256), cgrd((wn + 255) / 256);
  cast_f32_to_f16<<<cgrd, cblk, 0, stream>>>(Wq,  WqH, wn);
  cast_f32_to_f16<<<cgrd, cblk, 0, stream>>>(Wk,  WkH, wn);
  cast_f32_to_f16<<<cgrd, cblk, 0, stream>>>(Wv,  WvH, wn);
  cast_f32_to_f16<<<cgrd, cblk, 0, stream>>>(fc1, W1H, wn);
  cast_f32_to_f16<<<cgrd, cblk, 0, stream>>>(fc2, W2H, wn);

  // LN + projection (XH buffer reused sequentially; stream-ordered)
  ln_cast    <<<ROWS / 8, 256, 0, stream>>>(q, ln1g, ln1b, XH);
  gemm_f16<0><<<512, 256, 0, stream>>>(XH, WqH, (void*)QH, nullptr);
  ln_cast    <<<ROWS / 8, 256, 0, stream>>>(k, ln2g, ln2b, XH);
  gemm_f16<0><<<512, 256, 0, stream>>>(XH, WkH, (void*)KH, nullptr);
  ln_cast    <<<ROWS / 8, 256, 0, stream>>>(v, ln3g, ln3b, XH);
  gemm_f16<1><<<512, 256, 0, stream>>>(XH, WvH, (void*)VH, VT);

  // attention
  scores_k <<<128 * 32, 256, 0, stream>>>(QH, KH, attn);
  softmax_k<<<(128 * SEQ) / 8, 256, 0, stream>>>(attn);
  pv_k     <<<512, 256, 0, stream>>>(attn, VT, OH);

  // output projections
  gemm_f16<2><<<512, 256, 0, stream>>>(OH, W1H, (void*)out_dyn,  nullptr);
  gemm_f16<2><<<512, 256, 0, stream>>>(VH, W2H, (void*)out_stat, nullptr);

  (void)in_sizes; (void)n_in; (void)out_size; (void)ws_size;
}
